// LorentzTemporalBlock_2293512536885
// MI455X (gfx1250) — matrix operand order repaired and checked by reading, not compiled, not run on hardware
//
#include <hip/hip_runtime.h>

// ---------------------------------------------------------------------------
// Feature detection: CDNA5 async global->LDS copies (compile-safe fallback)
// ---------------------------------------------------------------------------
#if defined(__gfx1250__) && __has_builtin(__builtin_amdgcn_global_load_async_to_lds_b128) && __has_builtin(__builtin_amdgcn_s_wait_asynccnt)
#define HAVE_ASYNC 1
#else
#define HAVE_ASYNC 0
#endif

// ---------------------------------------------------------------------------
// Types / helpers
// ---------------------------------------------------------------------------
typedef __attribute__((ext_vector_type(16))) __bf16 v16bf;
typedef __attribute__((ext_vector_type(8)))  float  v8f;
typedef int v4i_vec __attribute__((vector_size(16)));   // matches builtin param

union Frag {
  v16bf v;
  unsigned short h[16];
  uint4 q[2];
};

__device__ __forceinline__ unsigned short f2bfbits(float f) {
  union { float f; unsigned u; } c; c.f = f;
  unsigned u = c.u;
  u += 0x7FFFu + ((u >> 16) & 1u);            // round-to-nearest-even
  return (unsigned short)(u >> 16);
}
__device__ __forceinline__ float bf2f(unsigned short h) {
  union { unsigned u; float f; } c; c.u = ((unsigned)h) << 16;
  return c.f;
}
__device__ __forceinline__ v8f vzero8() {
  v8f z;
#pragma unroll
  for (int e = 0; e < 8; ++e) z[e] = 0.0f;
  return z;
}
__device__ __forceinline__ v8f wmma_bf16(const Frag& a, const Frag& b, v8f c) {
  return __builtin_amdgcn_wmma_f32_16x16x32_bf16(
      false, a.v, false, b.v, (short)0, c, false, false);
}
__device__ __forceinline__ float wred_sum(float v) {
#pragma unroll
  for (int off = 16; off > 0; off >>= 1) v += __shfl_xor(v, off, 32);
  return v;
}

#if HAVE_ASYNC
__device__ __forceinline__ void async_cp16(const void* g, void* l) {
  __builtin_amdgcn_global_load_async_to_lds_b128(
      (__attribute__((address_space(1))) v4i_vec*)g,
      (__attribute__((address_space(3))) v4i_vec*)l, 0, 0);
}
#endif

// Problem constants
constexpr int Bc = 8, Tc = 512, Jc = 26, Dc = 256, Hc = 8, HIDc = 1024;
constexpr int NR = Bc * Tc * Jc;   // 106496 rows (multiple of 128)

// ---------------------------------------------------------------------------
// One-shot weight convert+transpose: Wt[n*K+k] = bf16(W[k*N+n])
// ---------------------------------------------------------------------------
__global__ __launch_bounds__(256) void conv_wt(const float* __restrict__ W,
                                               unsigned short* __restrict__ Wt,
                                               int K, int N) {
  int idx = blockIdx.x * 256 + threadIdx.x;
  int k = idx / N, n = idx % N;
  Wt[(size_t)n * K + k] = f2bfbits(W[idx]);
}

// ---------------------------------------------------------------------------
// WMMA GEMM: C[M,N] = act(A[M,K](bf16) * W[K,N] + bias), W given transposed
// (Wt[n][k], bf16). Block tile 128x128, BK=32; 8 waves, wave tile 32x64
// (2x4 WMMA tiles). Async double-buffered staging when available.
// ACT: 0=none, 1=elu+1, 2=tanh-gelu. OUTF32: 0 -> bf16 out, 1 -> f32 out.
// ---------------------------------------------------------------------------
template <int ACT, int OUTF32>
__global__ __launch_bounds__(256) void gemm_bf16(
    const unsigned short* __restrict__ A, const unsigned short* __restrict__ Wt,
    const float* __restrict__ bias, void* __restrict__ Cout,
    int M, int N, int K) {
#if HAVE_ASYNC
  __shared__ __align__(16) unsigned short sA[2][128 * 40];
  __shared__ __align__(16) unsigned short sB[2][128 * 40];
#else
  __shared__ __align__(16) unsigned short sA[1][128 * 40];
  __shared__ __align__(16) unsigned short sB[1][128 * 40];
#endif
  const int tid  = threadIdx.x;
  const int nb   = N >> 7;
  const int bm   = (int)(blockIdx.x / nb) * 128;
  const int bn   = (int)(blockIdx.x % nb) * 128;
  const int lane = tid & 31, wave = tid >> 5;
  const int wm   = (wave >> 1) * 32;   // 0,32,64,96
  const int wn   = (wave & 1) * 64;    // 0,64
  const int mr   = lane & 15, kb8 = (lane >> 4) * 8, kb16 = (lane >> 4) * 16;

  v8f acc[2][4];
#pragma unroll
  for (int i = 0; i < 2; ++i)
#pragma unroll
    for (int j = 0; j < 4; ++j) acc[i][j] = vzero8();

  const int nk = K >> 5;

#if HAVE_ASYNC
  // issue 4 async 16B copies per thread for tile `kt` into buffer `buf`
  auto stage = [&](int buf, int kt) {
#pragma unroll
    for (int it = 0; it < 2; ++it) {
      int lin = it * 256 + tid;
      int row = lin >> 2, seg = lin & 3;
      async_cp16(A  + (size_t)(bm + row) * K + kt + seg * 8,
                 &sA[buf][row * 40 + seg * 8]);
      async_cp16(Wt + (size_t)(bn + row) * K + kt + seg * 8,
                 &sB[buf][row * 40 + seg * 8]);
    }
  };
  stage(0, 0);
#endif

  for (int ki = 0; ki < nk; ++ki) {
    const int kt = ki << 5;
#if HAVE_ASYNC
    const int buf = ki & 1;
    if (ki + 1 < nk) {
      stage(buf ^ 1, kt + 32);
      __builtin_amdgcn_s_wait_asynccnt(4);  // previous tile's 4 copies done
    } else {
      __builtin_amdgcn_s_wait_asynccnt(0);
    }
    __syncthreads();
#else
    const int buf = 0;
#pragma unroll
    for (int it = 0; it < 2; ++it) {
      int lin = it * 256 + tid;
      int row = lin >> 2, seg = lin & 3;
      uint4 av = *reinterpret_cast<const uint4*>(
          A + (size_t)(bm + row) * K + kt + seg * 8);
      uint4 bv = *reinterpret_cast<const uint4*>(
          Wt + (size_t)(bn + row) * K + kt + seg * 8);
      *reinterpret_cast<uint4*>(&sA[0][row * 40 + seg * 8]) = av;
      *reinterpret_cast<uint4*>(&sB[0][row * 40 + seg * 8]) = bv;
    }
    __syncthreads();
#endif

    // fragments: two b128 LDS loads each (ISA 16-bit A/B VGPR layout)
    Frag fa[2], fb[4];
#pragma unroll
    for (int i = 0; i < 2; ++i) {
      int base = (wm + i * 16 + mr) * 40;
      fa[i].q[0] = *reinterpret_cast<const uint4*>(&sA[buf][base + kb8]);
      fa[i].q[1] = *reinterpret_cast<const uint4*>(&sA[buf][base + 16 + kb8]);
    }
#pragma unroll
    for (int jt = 0; jt < 4; ++jt) {
      int base = (wn + jt * 16 + mr) * 40;
      fb[jt].q[0] = *reinterpret_cast<const uint4*>(&sB[buf][base + kb8]);
      fb[jt].q[1] = *reinterpret_cast<const uint4*>(&sB[buf][base + 16 + kb8]);
    }
#pragma unroll
    for (int i = 0; i < 2; ++i)
#pragma unroll
      for (int jt = 0; jt < 4; ++jt)
        acc[i][jt] = wmma_bf16(fa[i], fb[jt], acc[i][jt]);
    __syncthreads();
  }

  // epilogue: bias + activation + store (C layout: VGPR r -> M=r | r+8)
#pragma unroll
  for (int i = 0; i < 2; ++i)
#pragma unroll
    for (int jt = 0; jt < 4; ++jt) {
      int gcol = bn + wn + jt * 16 + (lane & 15);
      float bv = bias[gcol];
#pragma unroll
      for (int r = 0; r < 8; ++r) {
        int m_l = r + ((lane >> 4) << 3);
        size_t grow = (size_t)(bm + wm + i * 16 + m_l);
        float val = acc[i][jt][r] + bv;
        if (ACT == 1) {
          val = (val > 0.0f) ? (val + 1.0f) : expf(val);   // elu(x)+1
        } else if (ACT == 2) {
          float x3 = val * val * val;                      // tanh gelu
          val = 0.5f * val *
                (1.0f + tanhf(0.7978845608028654f * (val + 0.044715f * x3)));
        }
        if (OUTF32)
          ((float*)Cout)[grow * N + gcol] = val;
        else
          ((unsigned short*)Cout)[grow * N + gcol] = f2bfbits(val);
      }
    }
}

// ---------------------------------------------------------------------------
// Linear attention per (b,j,h): kv = pk^T @ v over T, ksum = sum_t pk,
// num = pq @ kv, den = pq . ksum + eps ; out = num/den (bf16).
// pk/v are transposed to [dh][t] during LDS staging so that all WMMA
// fragments are contiguous b128 LDS loads.
// ---------------------------------------------------------------------------
__global__ __launch_bounds__(256) void attn_linear(
    const unsigned short* __restrict__ Qb, const unsigned short* __restrict__ Kb,
    const unsigned short* __restrict__ Vb, unsigned short* __restrict__ Ob) {
  constexpr int TS = 264;  // padded t-stride (bf16 elems), 528B = 33*16B
  __shared__ __align__(16) unsigned short sKT[32 * TS];  // pk^T chunk [dh][t]
  __shared__ __align__(16) unsigned short sVT[32 * TS];  // v^T  chunk [dh][t]
  __shared__ __align__(16) unsigned short sX[256 * 32];  // pq chunk [t][dh]
  __shared__ __align__(16) unsigned short sKVT[32 * 40]; // kv^T (bf16)
  __shared__ float sKV[32 * 32];
  __shared__ float sKS[32];
  __shared__ float sDEN[256];

  const int blk = blockIdx.x;
  const int b = blk / (Jc * Hc);
  const int rem = blk % (Jc * Hc);
  const int j = rem / Hc, h = rem % Hc;
  const int tid = threadIdx.x, lane = tid & 31, wave = tid >> 5;

  for (int i = tid; i < 1024; i += 256) sKV[i] = 0.0f;
  if (tid < 32) sKS[tid] = 0.0f;
  __syncthreads();

  const size_t rowBase = ((size_t)b * Tc) * Jc + j;  // row(t) = rowBase + t*J
  const int colOfs = h * 32;

  v8f pacc[2][2];
#pragma unroll
  for (int i = 0; i < 2; ++i)
#pragma unroll
    for (int jt = 0; jt < 2; ++jt) pacc[i][jt] = vzero8();
  float ksPart = 0.0f;

  const int mr = lane & 15, kb8 = (lane >> 4) * 8, kb16 = (lane >> 4) * 16;

  // ---- Phase 1: accumulate kv (32x32) and ksum over T in 2 chunks of 256 ---
  for (int c = 0; c < 2; ++c) {
    const int tg0 = c * 256;
    // stage pk,v transposed: coalesced b128 global loads, b16 LDS scatter
#pragma unroll
    for (int it = 0; it < 4; ++it) {
      int lin = it * 256 + tid;
      int tl = lin >> 2, seg = lin & 3;
      size_t gofs = (rowBase + (size_t)(tg0 + tl) * Jc) * Dc + colOfs + seg * 8;
      uint4 kq = *reinterpret_cast<const uint4*>(Kb + gofs);
      uint4 vq = *reinterpret_cast<const uint4*>(Vb + gofs);
      unsigned kw[4] = {kq.x, kq.y, kq.z, kq.w};
      unsigned vw[4] = {vq.x, vq.y, vq.z, vq.w};
#pragma unroll
      for (int e = 0; e < 4; ++e) {
        int d = seg * 8 + e * 2;
        sKT[(d    ) * TS + tl] = (unsigned short)(kw[e] & 0xffff);
        sKT[(d + 1) * TS + tl] = (unsigned short)(kw[e] >> 16);
        sVT[(d    ) * TS + tl] = (unsigned short)(vw[e] & 0xffff);
        sVT[(d + 1) * TS + tl] = (unsigned short)(vw[e] >> 16);
      }
    }
    __syncthreads();

    const int t0 = wave * 32;  // this wave's 32 timesteps of the chunk
    Frag fa[2], fb[2];
#pragma unroll
    for (int i = 0; i < 2; ++i) {        // A = pk^T : rows are dk, cols t
      int base = (i * 16 + mr) * TS + t0;
      fa[i].q[0] = *reinterpret_cast<const uint4*>(&sKT[base + kb8]);
      fa[i].q[1] = *reinterpret_cast<const uint4*>(&sKT[base + 16 + kb8]);
    }
#pragma unroll
    for (int jt = 0; jt < 2; ++jt) {     // B = v : col n from sVT row n
      int base = (jt * 16 + mr) * TS + t0;
      fb[jt].q[0] = *reinterpret_cast<const uint4*>(&sVT[base + kb8]);
      fb[jt].q[1] = *reinterpret_cast<const uint4*>(&sVT[base + 16 + kb8]);
    }
#pragma unroll
    for (int i = 0; i < 2; ++i)
#pragma unroll
      for (int jt = 0; jt < 2; ++jt)
        pacc[i][jt] = wmma_bf16(fa[i], fb[jt], pacc[i][jt]);

    {  // ksum partial: row d of sKT, 32 t-values, vectorized
      int d = tid & 31, tq = tid >> 5;
      const uint4* rp =
          reinterpret_cast<const uint4*>(&sKT[d * TS + tq * 32]);
#pragma unroll
      for (int c4 = 0; c4 < 4; ++c4) {
        uint4 qv = rp[c4];
        unsigned w[4] = {qv.x, qv.y, qv.z, qv.w};
#pragma unroll
        for (int e = 0; e < 4; ++e) {
          ksPart += bf2f((unsigned short)(w[e] & 0xffff));
          ksPart += bf2f((unsigned short)(w[e] >> 16));
        }
      }
    }
    __syncthreads();
  }

  // combine wave partials in LDS (ds_add_f32)
#pragma unroll
  for (int i = 0; i < 2; ++i)
#pragma unroll
    for (int jt = 0; jt < 2; ++jt)
#pragma unroll
      for (int r = 0; r < 8; ++r) {
        int mrow = i * 16 + r + ((lane >> 4) << 3);
        int ncol = jt * 16 + (lane & 15);
        atomicAdd(&sKV[mrow * 32 + ncol], pacc[i][jt][r]);
      }
  atomicAdd(&sKS[tid & 31], ksPart);
  __syncthreads();

  // convert + transpose kv into bf16: sKVT[n][k] = bf16(kv[k][n])
#pragma unroll
  for (int i = 0; i < 4; ++i) {
    int idx = tid + i * 256;
    int k = idx >> 5, n = idx & 31;
    sKVT[n * 40 + k] = f2bfbits(sKV[k * 32 + n]);
  }
  __syncthreads();

  // ---- Phase 2: num = pq @ kv, den, write out -----------------------------
  for (int c = 0; c < 2; ++c) {
    const int tg0 = c * 256;
#pragma unroll
    for (int it = 0; it < 4; ++it) {
      int lin = it * 256 + tid;
      int tl = lin >> 2, seg = lin & 3;
      size_t gofs = (rowBase + (size_t)(tg0 + tl) * Jc) * Dc + colOfs + seg * 8;
#if HAVE_ASYNC
      async_cp16(Qb + gofs, &sX[tl * 32 + seg * 8]);
#else
      *reinterpret_cast<uint4*>(&sX[tl * 32 + seg * 8]) =
          *reinterpret_cast<const uint4*>(Qb + gofs);
#endif
    }
#if HAVE_ASYNC
    __builtin_amdgcn_s_wait_asynccnt(0);
#endif
    __syncthreads();
    {  // den[t] = pq[t,:] . ksum  (vectorized LDS reads)
      float s = 0.0f;
      const uint4* rp = reinterpret_cast<const uint4*>(&sX[tid * 32]);
#pragma unroll
      for (int c4 = 0; c4 < 4; ++c4) {
        uint4 qv = rp[c4];
        unsigned w[4] = {qv.x, qv.y, qv.z, qv.w};
#pragma unroll
        for (int e = 0; e < 4; ++e) {
          s += bf2f((unsigned short)(w[e] & 0xffff)) * sKS[c4 * 8 + e * 2];
          s += bf2f((unsigned short)(w[e] >> 16))    * sKS[c4 * 8 + e * 2 + 1];
        }
      }
      sDEN[tid] = s + 1e-6f;
    }
    __syncthreads();

#pragma unroll
    for (int mi = 0; mi < 2; ++mi) {
      const int t0 = (wave * 2 + mi) * 16;
      Frag fa;
      {
        int base = (t0 + mr) * 32;
        fa.q[0] = *reinterpret_cast<const uint4*>(&sX[base + kb8]);
        fa.q[1] = *reinterpret_cast<const uint4*>(&sX[base + 16 + kb8]);
      }
#pragma unroll
      for (int jt = 0; jt < 2; ++jt) {
        Frag fb;
        int base = (jt * 16 + mr) * 40;
        fb.q[0] = *reinterpret_cast<const uint4*>(&sKVT[base + kb8]);
        fb.q[1] = *reinterpret_cast<const uint4*>(&sKVT[base + 16 + kb8]);
        v8f acc = wmma_bf16(fa, fb, vzero8());
#pragma unroll
        for (int r = 0; r < 8; ++r) {
          int tl = t0 + r + ((lane >> 4) << 3);
          int n = jt * 16 + (lane & 15);
          float val = acc[r] / sDEN[tl];
          size_t gofs = (rowBase + (size_t)(tg0 + tl) * Jc) * Dc + colOfs + n;
          Ob[gofs] = f2bfbits(val);
        }
      }
    }
    __syncthreads();
  }
}

// ---------------------------------------------------------------------------
// Pointwise kernels: one wave32 per row, 8 elems/lane (stride-32 coalesced)
// ---------------------------------------------------------------------------
__global__ __launch_bounds__(256) void pw_pre(
    const float* __restrict__ x, const float* __restrict__ g1,
    const float* __restrict__ b1, unsigned short* __restrict__ tout) {
  const int row = blockIdx.x * 8 + (threadIdx.x >> 5);
  const int lane = threadIdx.x & 31;
  const float* xr = x + (size_t)row * 257;
  const float x0 = xr[0];
  float xs[8];
#pragma unroll
  for (int i = 0; i < 8; ++i) xs[i] = xr[1 + lane + i * 32];
  float ss = 0.0f;
#pragma unroll
  for (int i = 0; i < 8; ++i) ss += xs[i] * xs[i];
  ss = wred_sum(ss);
  float nx = fmaxf(sqrtf(ss), 1e-6f);
  float sc = acoshf(fmaxf(x0, 1.0f + 1e-7f)) / nx;
  float u[8];
  float s1 = 0.0f, s2 = 0.0f;
#pragma unroll
  for (int i = 0; i < 8; ++i) { u[i] = xs[i] * sc; s1 += u[i]; s2 += u[i] * u[i]; }
  float mean = wred_sum(s1) * (1.0f / 256.0f);
  float msq  = wred_sum(s2) * (1.0f / 256.0f);
  float rstd = rsqrtf(msq - mean * mean + 1e-5f);
  float un2 = 0.0f;
#pragma unroll
  for (int i = 0; i < 8; ++i) {
    int idx = lane + i * 32;
    u[i] = (u[i] - mean) * rstd * g1[idx] + b1[idx];
    un2 += u[i] * u[i];
  }
  float nu = fmaxf(sqrtf(wred_sum(un2)), 1e-6f);
  float ch = coshf(nu);
  float shn = sinhf(nu) / nu;
  float xs2[8];
  float n2acc = 0.0f;
#pragma unroll
  for (int i = 0; i < 8; ++i) { xs2[i] = shn * u[i]; n2acc += xs2[i] * xs2[i]; }
  float n2 = fmaxf(sqrtf(wred_sum(n2acc)), 1e-6f);
  float sc2 = acoshf(fmaxf(ch, 1.0f + 1e-7f)) / n2;
#pragma unroll
  for (int i = 0; i < 8; ++i)
    tout[(size_t)row * 256 + lane + i * 32] = f2bfbits(xs2[i] * sc2);
}

__global__ __launch_bounds__(256) void pw_mid(
    const float* __restrict__ x, const float* __restrict__ o,
    const float* __restrict__ g2, const float* __restrict__ b2,
    unsigned short* __restrict__ u2out) {
  const int row = blockIdx.x * 8 + (threadIdx.x >> 5);
  const int lane = threadIdx.x & 31;
  const float* xr = x + (size_t)row * 257;
  const float* orow = o + (size_t)row * 256;
  float ov[8];
  float so = 0.0f;
#pragma unroll
  for (int i = 0; i < 8; ++i) { ov[i] = orow[lane + i * 32]; so += ov[i] * ov[i]; }
  float no = fmaxf(sqrtf(wred_sum(so)), 1e-6f);
  float z0 = coshf(no);
  float zsc = sinhf(no) / no;
  float m0 = xr[0] + z0;
  float ms[8];
  float smsq = 0.0f;
#pragma unroll
  for (int i = 0; i < 8; ++i) {
    ms[i] = xr[1 + lane + i * 32] + zsc * ov[i];
    smsq += ms[i] * ms[i];
  }
  float mink = -m0 * m0 + wred_sum(smsq);
  float dn = sqrtf(fmaxf(-mink, 1e-6f));
  float h0 = m0 / dn;
  float hs[8];
  float shs = 0.0f;
#pragma unroll
  for (int i = 0; i < 8; ++i) { hs[i] = ms[i] / dn; shs += hs[i] * hs[i]; }
  float nh = fmaxf(sqrtf(wred_sum(shs)), 1e-6f);
  float sch = acoshf(fmaxf(h0, 1.0f + 1e-7f)) / nh;
  float v[8];
  float s1 = 0.0f, s2 = 0.0f;
#pragma unroll
  for (int i = 0; i < 8; ++i) { v[i] = hs[i] * sch; s1 += v[i]; s2 += v[i] * v[i]; }
  float mean = wred_sum(s1) * (1.0f / 256.0f);
  float msq  = wred_sum(s2) * (1.0f / 256.0f);
  float rstd = rsqrtf(msq - mean * mean + 1e-5f);
#pragma unroll
  for (int i = 0; i < 8; ++i) {
    int idx = lane + i * 32;
    float val = (v[i] - mean) * rstd * g2[idx] + b2[idx];
    u2out[(size_t)row * 256 + idx] = f2bfbits(val);
  }
}

__global__ __launch_bounds__(256) void pw_out(
    const float* __restrict__ f, float* __restrict__ out) {
  const int row = blockIdx.x * 8 + (threadIdx.x >> 5);
  const int lane = threadIdx.x & 31;
  const float* fr = f + (size_t)row * 256;
  float fv[8];
  float sf = 0.0f;
#pragma unroll
  for (int i = 0; i < 8; ++i) { fv[i] = fr[lane + i * 32]; sf += fv[i] * fv[i]; }
  float n = fmaxf(sqrtf(wred_sum(sf)), 1e-6f);
  float c = coshf(n);
  float s = sinhf(n) / n;
  float* orow = out + (size_t)row * 257;
  if (lane == 0) orow[0] = c;
#pragma unroll
  for (int i = 0; i < 8; ++i) orow[1 + lane + i * 32] = s * fv[i];
}

// ---------------------------------------------------------------------------
// Launch
// ---------------------------------------------------------------------------
extern "C" void kernel_launch(void* const* d_in, const int* in_sizes, int n_in,
                              void* d_out, int out_size, void* d_ws,
                              size_t ws_size, hipStream_t stream) {
  const float* x     = (const float*)d_in[0];
  const float* g1    = (const float*)d_in[1];
  const float* beta1 = (const float*)d_in[2];
  const float* Wq    = (const float*)d_in[3];
  const float* Wk    = (const float*)d_in[4];
  const float* Wv    = (const float*)d_in[5];
  const float* Wo    = (const float*)d_in[6];
  const float* pbq   = (const float*)d_in[7];
  const float* pbk   = (const float*)d_in[8];
  const float* pbv   = (const float*)d_in[9];
  const float* pbo   = (const float*)d_in[10];
  const float* g2    = (const float*)d_in[11];
  const float* beta2 = (const float*)d_in[12];
  const float* W1    = (const float*)d_in[13];
  const float* pbf1  = (const float*)d_in[14];
  const float* W2    = (const float*)d_in[15];
  const float* pbf2  = (const float*)d_in[16];
  float* out = (float*)d_out;

  unsigned char* ws = (unsigned char*)d_ws;
  const size_t szA = (size_t)NR * Dc * 2;  // one N x 256 bf16 buffer
  unsigned short* wt  = (unsigned short*)(ws);            // t, later attn-out
  unsigned short* wq  = (unsigned short*)(ws + szA);      // q, later u2
  unsigned short* wk  = (unsigned short*)(ws + 2 * szA);  // k
  unsigned short* wv  = (unsigned short*)(ws + 3 * szA);  // v
  unsigned short* wff = (unsigned short*)(ws + 4 * szA);  // ffn hidden (4*szA)
  float* wo_f32 = (float*)(ws + 2 * szA);  // aliases k+v (dead after attn)
  float* wf_f32 = (float*)(ws + 2 * szA);  // aliases o   (dead after pw_mid)
  unsigned short* wu2 = wq;                // aliases q   (dead after attn)
  // transposed bf16 weights (1.5 MB) after the ffn buffer
  unsigned short* wtq = (unsigned short*)(ws + 8 * szA);
  unsigned short* wtk = wtq + 65536;
  unsigned short* wtv = wtk + 65536;
  unsigned short* wto = wtv + 65536;
  unsigned short* wt1 = wto + 65536;            // 1024 x 256
  unsigned short* wt2 = wt1 + 262144;           // 256 x 1024

  const int pwBlocks = NR / 8;                  // 13312
  const int gB256 = (NR / 128) * (Dc / 128);    // 1664
  const int gB1k  = (NR / 128) * (HIDc / 128);  // 6656

  // one-shot weight transpose+convert to bf16
  conv_wt<<<256, 256, 0, stream>>>(Wq, wtq, Dc, Dc);
  conv_wt<<<256, 256, 0, stream>>>(Wk, wtk, Dc, Dc);
  conv_wt<<<256, 256, 0, stream>>>(Wv, wtv, Dc, Dc);
  conv_wt<<<256, 256, 0, stream>>>(Wo, wto, Dc, Dc);
  conv_wt<<<1024, 256, 0, stream>>>(W1, wt1, Dc, HIDc);
  conv_wt<<<1024, 256, 0, stream>>>(W2, wt2, HIDc, Dc);

  // t = ln(logmap0(x)) -> expmap0 -> logmap0 (bf16)
  pw_pre<<<pwBlocks, 256, 0, stream>>>(x, g1, beta1, wt);
  // q/k/v GEMMs (elu+1 fused into q,k)
  gemm_bf16<1, 0><<<gB256, 256, 0, stream>>>(wt, wtq, pbq, wq, NR, Dc, Dc);
  gemm_bf16<1, 0><<<gB256, 256, 0, stream>>>(wt, wtk, pbk, wk, NR, Dc, Dc);
  gemm_bf16<0, 0><<<gB256, 256, 0, stream>>>(wt, wtv, pbv, wv, NR, Dc, Dc);
  // linear attention, writes attn-out into wt (bf16)
  attn_linear<<<Bc * Jc * Hc, 256, 0, stream>>>(wq, wk, wv, wt);
  // o = attn @ Wo + bo (f32)
  gemm_bf16<0, 1><<<gB256, 256, 0, stream>>>(wt, wto, pbo, (void*)wo_f32, NR, Dc, Dc);
  // u2 = ln(logmap0(lorentz_residual(x, expmap0(o)))) (bf16)
  pw_mid<<<pwBlocks, 256, 0, stream>>>(x, wo_f32, g2, beta2, wu2);
  // FFN
  gemm_bf16<2, 0><<<gB1k, 256, 0, stream>>>(wu2, wt1, pbf1, wff, NR, HIDc, Dc);
  gemm_bf16<0, 1><<<gB256, 256, 0, stream>>>(wff, wt2, pbf2, (void*)wf_f32, NR, Dc, HIDc);
  // out = expmap0(f)
  pw_out<<<pwBlocks, 256, 0, stream>>>(wf_f32, out);
}
